// CCHead_65816078844147
// MI455X (gfx1250) — matrix operand order repaired and checked by reading, not compiled, not run on hardware
//
#include <hip/hip_runtime.h>

typedef __attribute__((ext_vector_type(16))) __bf16 v16bf;
typedef __attribute__((ext_vector_type(8)))  float  v8f;

union FragB { v16bf v; uint4 q[2]; };

#define NPIX 32768   // 8*64*64

__device__ __forceinline__ unsigned short f2bf(float f) {
    union { float f; unsigned u; } x; x.f = f;
    unsigned r = x.u + 0x7FFFu + ((x.u >> 16) & 1u);   // RNE
    return (unsigned short)(r >> 16);
}
__device__ __forceinline__ float bf2f(unsigned short h) {
    union { unsigned u; float f; } x; x.u = ((unsigned)h) << 16;
    return x.f;
}

// CDNA5 async global->LDS copy (ASYNCcnt-tracked, no VGPR staging)
__device__ __forceinline__ void async_ld_b128(unsigned lds_off, const void* gptr) {
    asm volatile("global_load_async_to_lds_b128 %0, %1, off"
                 :: "v"(lds_off), "v"(gptr) : "memory");
}
__device__ __forceinline__ void wait_async0() {
    asm volatile("s_wait_asynccnt 0x0" ::: "memory");
}

// ---------------------------------------------------------------------------
// NCHW fp32 -> NHWC bf16 tiled transpose (x: C=2048)
// ---------------------------------------------------------------------------
__global__ __launch_bounds__(256) void cvt_nhwc(const float* __restrict__ src,
                                                unsigned short* __restrict__ dst, int C) {
    __shared__ float tile[32][33];
    const int tx = threadIdx.x & 31, ty = threadIdx.x >> 5;
    const int n = blockIdx.z;
#pragma unroll
    for (int rr = 0; rr < 4; ++rr) {
        int c  = blockIdx.x * 32 + ty + rr * 8;
        int hw = blockIdx.y * 32 + tx;
        tile[ty + rr * 8][tx] = src[((size_t)(n * C + c)) * 4096 + hw];
    }
    __syncthreads();
#pragma unroll
    for (int rr = 0; rr < 4; ++rr) {
        int hw = blockIdx.y * 32 + ty + rr * 8;
        int c  = blockIdx.x * 32 + tx;
        dst[((size_t)(n * 4096 + hw)) * C + c] = f2bf(tile[tx][ty + rr * 8]);
    }
}

// ---------------------------------------------------------------------------
// Weight transform: (Cout, Ctot, KH, KW) fp32 -> [Cout][t*Ctot + c] bf16
// ---------------------------------------------------------------------------
__global__ void wtrans(const float* __restrict__ src, unsigned short* __restrict__ dst,
                       int Cout, int Ctot, int KHW) {
    int idx = blockIdx.x * blockDim.x + threadIdx.x;
    int total = Cout * KHW * Ctot;
    if (idx >= total) return;
    int co  = idx / (KHW * Ctot);
    int rem = idx - co * (KHW * Ctot);
    int t   = rem / Ctot;
    int c   = rem - t * Ctot;
    dst[idx] = f2bf(src[((size_t)(co * Ctot + c)) * KHW + t]);
}

__global__ void bn_prep(const float* g, const float* b, const float* m, const float* v,
                        float* scale, float* bias, int C) {
    int i = blockIdx.x * blockDim.x + threadIdx.x;
    if (i >= C) return;
    float s = g[i] * rsqrtf(v[i] + 1e-5f);
    scale[i] = s; bias[i] = b[i] - m[i] * s;
}
__global__ void bias_prep(const float* b, float* scale, float* bias, int C) {
    int i = blockIdx.x * blockDim.x + threadIdx.x;
    if (i >= C) return;
    scale[i] = 1.0f; bias[i] = b[i];
}

// ---------------------------------------------------------------------------
// Implicit-GEMM conv (NHWC bf16 in, bf16 WMMA, f32 accum, fused scale/bias/relu)
// Double-buffered LDS tiles filled by async global->LDS; concat of two inputs
// along channels (C0 then C1, both %32==0).
// Tile: 128(M=Cout) x 128(N=pixels), K-step 32.  256 thr = 8 waves, each 64x32.
// ---------------------------------------------------------------------------
__global__ __launch_bounds__(256)
void conv_gemm(const unsigned short* __restrict__ X0, int C0,
               const unsigned short* __restrict__ X1, int C1,
               const unsigned short* __restrict__ Wt, int Cout, int KH, int KW,
               const float* __restrict__ scale, const float* __restrict__ bias, int relu,
               unsigned short* __restrict__ outH, float* __restrict__ outN) {
    const int Ctot = C0 + C1;
    const int nc32 = Ctot >> 5;
    const int S    = KH * KW * nc32;
    const int Kdim = KH * KW * Ctot;
    const int m0 = blockIdx.x * 128;
    const int n0 = blockIdx.y * 128;
    const int tid = threadIdx.x;
    const int lane = tid & 31, wave = tid >> 5;
    const int wm = wave & 1, wn = wave >> 1;
    const int lr = lane & 15;
    const int kg = (lane >> 4) << 3;    // A frag: 0 / 8
    const int kb = (lane >> 4) << 4;    // unused (B frag also uses kg): 0 / 16
    const int msel = kg;

    // [0 .. 10240): As[2][128*40]   [10240 .. 20480): Bs[2][128*40]
    __shared__ unsigned short smem[20480];
    const unsigned smem0 = (unsigned)(size_t)(&smem[0]);

    auto issue = [&](int step, int buf) {
        int t  = step / nc32;
        int c0 = (step - t * nc32) << 5;
        int dy = (t / KW) - (KH >> 1);
        int dx = (t % KW) - (KW >> 1);
#pragma unroll
        for (int s = 0; s < 2; ++s) {
            int cid = tid + s * 256;
            int row = cid >> 2, part = cid & 3;
            unsigned lofs = (unsigned)(row * 40 + part * 8) * 2u;
            // ---- A: weights ----
            int co = m0 + row;
            if (co < Cout) {
                const unsigned short* gp = Wt + (size_t)co * Kdim + t * Ctot + c0 + part * 8;
                async_ld_b128(smem0 + (unsigned)(buf * 5120) * 2u + lofs, gp);
            } else {
                uint4 z; z.x = z.y = z.z = z.w = 0u;
                *(uint4*)(&smem[buf * 5120 + row * 40 + part * 8]) = z;
            }
            // ---- B: im2col pixels ----
            int pix = n0 + row;
            int n = pix >> 12, hw = pix & 4095;
            int y = (hw >> 6) + dy, x = (hw & 63) + dx;
            if ((unsigned)y < 64u && (unsigned)x < 64u) {
                int c = c0 + part * 8;
                size_t poff = (size_t)((n * 64 + y) * 64 + x);
                const unsigned short* gp = (c < C0) ? (X0 + poff * C0 + c)
                                                    : (X1 + poff * C1 + (c - C0));
                async_ld_b128(smem0 + (unsigned)(10240 + buf * 5120) * 2u + lofs, gp);
            } else {
                uint4 z; z.x = z.y = z.z = z.w = 0u;
                *(uint4*)(&smem[10240 + buf * 5120 + row * 40 + part * 8]) = z;
            }
        }
    };

    v8f acc[4][2];
#pragma unroll
    for (int i = 0; i < 4; ++i)
#pragma unroll
        for (int j = 0; j < 2; ++j)
#pragma unroll
            for (int r = 0; r < 8; ++r) acc[i][j][r] = 0.0f;

    issue(0, 0);
    wait_async0();
    __syncthreads();

    for (int step = 0; step < S; ++step) {
        const int buf = step & 1;
        if (step + 1 < S) issue(step + 1, buf ^ 1);   // overlap with WMMA below

        const unsigned short* Asb = &smem[buf * 5120];
        const unsigned short* Bsb = &smem[10240 + buf * 5120];
        FragB a[4], b[2];
#pragma unroll
        for (int mt = 0; mt < 4; ++mt) {
            int m = wm * 64 + mt * 16 + lr;
            a[mt].q[0] = *(const uint4*)(&Asb[m * 40 + kg]);
            a[mt].q[1] = *(const uint4*)(&Asb[m * 40 + 16 + kg]);
        }
#pragma unroll
        for (int nt = 0; nt < 2; ++nt) {
            int n = wn * 32 + nt * 16 + lr;
            b[nt].q[0] = *(const uint4*)(&Bsb[n * 40 + kg]);
            b[nt].q[1] = *(const uint4*)(&Bsb[n * 40 + 16 + kg]);
        }
#pragma unroll
        for (int mt = 0; mt < 4; ++mt)
#pragma unroll
            for (int nt = 0; nt < 2; ++nt)
                acc[mt][nt] = __builtin_amdgcn_wmma_f32_16x16x32_bf16(
                    false, a[mt].v, false, b[nt].v, (short)0, acc[mt][nt], false, false);

        wait_async0();       // my async writes into buf^1 are in LDS
        __syncthreads();     // everyone's writes visible; safe to overwrite buf next step
    }

    // ---- epilogue ----
    if (outH) {
        // stage scaled bf16 tile [128 pix][128+8 co] in LDS, then coalesced b128 stores
        int cw = Cout - m0; if (cw > 128) cw = 128;
#pragma unroll
        for (int mt = 0; mt < 4; ++mt)
#pragma unroll
            for (int nt = 0; nt < 2; ++nt)
#pragma unroll
                for (int r = 0; r < 8; ++r) {
                    int col = wm * 64 + mt * 16 + r + msel;
                    int pl  = wn * 32 + nt * 16 + lr;
                    if (col < cw) {
                        float v = acc[mt][nt][r] * scale[m0 + col] + bias[m0 + col];
                        if (relu) v = fmaxf(v, 0.0f);
                        smem[pl * 136 + col] = f2bf(v);
                    }
                }
        __syncthreads();
        const int chunks = cw >> 3;                 // 8 halves per chunk
        for (int cid = tid; cid < 128 * chunks; cid += 256) {
            int row = cid / chunks, part = cid - row * chunks;
            uint4 val = *(const uint4*)(&smem[row * 136 + part * 8]);
            int pix = n0 + row;
            *(uint4*)(outH + (size_t)pix * Cout + m0 + part * 8) = val;
        }
    }
    if (outN) {
#pragma unroll
        for (int mt = 0; mt < 4; ++mt)
#pragma unroll
            for (int nt = 0; nt < 2; ++nt)
#pragma unroll
                for (int r = 0; r < 8; ++r) {
                    int co  = m0 + wm * 64 + mt * 16 + r + msel;
                    int pix = n0 + wn * 32 + nt * 16 + lr;
                    if (co < Cout) {
                        float v = acc[mt][nt][r] * scale[co] + bias[co];
                        if (relu) v = fmaxf(v, 0.0f);
                        int n = pix >> 12, hw = pix & 4095;
                        outN[((size_t)n * Cout + co) * 4096 + hw] = v;
                    }
                }
    }
}

// ---------------------------------------------------------------------------
// CCA scores: mode 0 -> eh over column w (diag mask), mode 1 -> ew over row h.
// 64x64 GEMM, K=64, per block (128 thr = 4 waves, each a 16-row strip).
// ---------------------------------------------------------------------------
__global__ __launch_bounds__(128)
void cca_scores(const unsigned short* __restrict__ q, const unsigned short* __restrict__ k,
                float* __restrict__ scores, int mode) {
    const int b = blockIdx.x >> 6;
    const int f = blockIdx.x & 63;            // w (mode 0) or h (mode 1)
    const int tid = threadIdx.x;
    __shared__ unsigned short Qs[64 * 72];
    __shared__ unsigned short Ks[64 * 72];

#pragma unroll
    for (int it = 0; it < 4; ++it) {
        int cid = tid + it * 128;             // 512 chunks of 8 halves
        int row = cid >> 3, part = cid & 7;
        size_t pix = mode ? (size_t)((b * 64 + f) * 64 + row)
                          : (size_t)((b * 64 + row) * 64 + f);
        *(uint4*)(&Qs[row * 72 + part * 8]) = *(const uint4*)(q + pix * 64 + part * 8);
    }
#pragma unroll
    for (int it = 0; it < 4; ++it) {
        int cid = tid + it * 128;
        int row = cid >> 3, part = cid & 7;
        size_t pix = mode ? (size_t)((b * 64 + f) * 64 + row)
                          : (size_t)((b * 64 + row) * 64 + f);
        *(uint4*)(&Ks[row * 72 + part * 8]) = *(const uint4*)(k + pix * 64 + part * 8);
    }
    __syncthreads();

    const int wave = tid >> 5, lane = tid & 31, lr = lane & 15;
    const int kg = (lane >> 4) << 3, kb = (lane >> 4) << 4, msel = kg;

    FragB a[2];
    const int m = wave * 16 + lr;
#pragma unroll
    for (int ks = 0; ks < 2; ++ks) {
        a[ks].q[0] = *(const uint4*)(&Qs[m * 72 + ks * 32 + kg]);
        a[ks].q[1] = *(const uint4*)(&Qs[m * 72 + ks * 32 + 16 + kg]);
    }
    v8f acc[4];
#pragma unroll
    for (int nt = 0; nt < 4; ++nt) {
#pragma unroll
        for (int r = 0; r < 8; ++r) acc[nt][r] = 0.0f;
#pragma unroll
        for (int ks = 0; ks < 2; ++ks) {
            FragB bf;
            int n = nt * 16 + lr;
            bf.q[0] = *(const uint4*)(&Ks[n * 72 + ks * 32 + kg]);
            bf.q[1] = *(const uint4*)(&Ks[n * 72 + ks * 32 + 16 + kg]);
            acc[nt] = __builtin_amdgcn_wmma_f32_16x16x32_bf16(
                false, a[ks].v, false, bf.v, (short)0, acc[nt], false, false);
        }
    }
#pragma unroll
    for (int nt = 0; nt < 4; ++nt)
#pragma unroll
        for (int r = 0; r < 8; ++r) {
            int i = wave * 16 + r + msel;
            int j = nt * 16 + lr;
            float v = acc[nt][r];
            if (mode == 0 && i == j) v = -1e30f;
            size_t pix = mode ? (size_t)((b * 64 + f) * 64 + i)
                              : (size_t)((b * 64 + i) * 64 + f);
            scores[pix * 128 + (mode ? 64 + j : j)] = v;
        }
}

// ---------------------------------------------------------------------------
// Softmax over 128 logits, one wave per position; split into attH/attW bf16.
// attH layout [b][w][h][j], attW layout [b][h][w][j].
// ---------------------------------------------------------------------------
__global__ __launch_bounds__(256)
void cca_softmax(const float* __restrict__ scores,
                 unsigned short* __restrict__ attH, unsigned short* __restrict__ attW) {
    const int lane = threadIdx.x & 31;
    const int p = blockIdx.x * 8 + (threadIdx.x >> 5);
    float4 v = *(const float4*)(scores + (size_t)p * 128 + lane * 4);
    float mx = fmaxf(fmaxf(v.x, v.y), fmaxf(v.z, v.w));
#pragma unroll
    for (int o = 16; o; o >>= 1) mx = fmaxf(mx, __shfl_xor(mx, o));
    float e0 = __expf(v.x - mx), e1 = __expf(v.y - mx),
          e2 = __expf(v.z - mx), e3 = __expf(v.w - mx);
    float sm = e0 + e1 + e2 + e3;
#pragma unroll
    for (int o = 16; o; o >>= 1) sm += __shfl_xor(sm, o);
    float inv = 1.0f / sm;
    int b = p >> 12, hw = p & 4095, h = hw >> 6, w = hw & 63;
    float ee[4] = {e0 * inv, e1 * inv, e2 * inv, e3 * inv};
#pragma unroll
    for (int kk = 0; kk < 4; ++kk) {
        int j = lane * 4 + kk;
        if (j < 64) attH[((size_t)((b * 64 + w) * 64 + h)) * 64 + j] = f2bf(ee[kk]);
        else        attW[((size_t)((b * 64 + h) * 64 + w)) * 64 + (j - 64)] = f2bf(ee[kk]);
    }
}

// ---------------------------------------------------------------------------
// CCA output GEMM: att(64x64) x v(64x512), v transposed into LDS [c][j].
// mode 0: out_h -> tmpH.   mode 1: out_w; fuse gamma*(tmpH+acc)+x -> oout bf16.
// ---------------------------------------------------------------------------
__global__ __launch_bounds__(256)
void cca_out(const unsigned short* __restrict__ att, const unsigned short* __restrict__ v,
             float* __restrict__ tmpH, const unsigned short* __restrict__ xin,
             const float* __restrict__ gamma, unsigned short* __restrict__ oout, int mode) {
    const int b = blockIdx.x >> 6;
    const int f = blockIdx.x & 63;            // w (mode 0) or h (mode 1)
    const int tid = threadIdx.x;
    __shared__ unsigned short As_[64 * 72];
    __shared__ unsigned short Vt[256 * 72];

    const unsigned short* abase = att + ((size_t)(b * 64 + f)) * 4096;
#pragma unroll
    for (int s = 0; s < 2; ++s) {
        int cid = tid + s * 256;
        int row = cid >> 3, part = cid & 7;
        *(uint4*)(&As_[row * 72 + part * 8]) = *(const uint4*)(abase + cid * 8);
    }

    const int wave = tid >> 5, lane = tid & 31, lr = lane & 15;
    const int wm = wave & 3, wvn = wave >> 2;
    const int kg = (lane >> 4) << 3, kb = (lane >> 4) << 4, msel = kg;
    const float g0 = mode ? gamma[0] : 0.0f;

    FragB a[2];
    for (int h2 = 0; h2 < 2; ++h2) {
        // load half of transposed v: c in [h2*256, h2*256+256)
        for (int idx = tid; idx < 16384; idx += 256) {
            int j = idx >> 8;
            int c = (idx & 255) + h2 * 256;
            size_t pix = mode ? (size_t)((b * 64 + f) * 64 + j)
                              : (size_t)((b * 64 + j) * 64 + f);
            Vt[(c - h2 * 256) * 72 + j] = v[pix * 512 + c];
        }
        __syncthreads();
        if (h2 == 0) {
            int m = wm * 16 + lr;
#pragma unroll
            for (int ks = 0; ks < 2; ++ks) {
                a[ks].q[0] = *(const uint4*)(&As_[m * 72 + ks * 32 + kg]);
                a[ks].q[1] = *(const uint4*)(&As_[m * 72 + ks * 32 + 16 + kg]);
            }
        }
#pragma unroll
        for (int nt = 0; nt < 8; ++nt) {
            int cl = wvn * 128 + nt * 16;       // local c within half
            v8f acc;
#pragma unroll
            for (int r = 0; r < 8; ++r) acc[r] = 0.0f;
#pragma unroll
            for (int ks = 0; ks < 2; ++ks) {
                FragB bf;
                int n = cl + lr;
                bf.q[0] = *(const uint4*)(&Vt[n * 72 + ks * 32 + kg]);
                bf.q[1] = *(const uint4*)(&Vt[n * 72 + ks * 32 + 16 + kg]);
                acc = __builtin_amdgcn_wmma_f32_16x16x32_bf16(
                    false, a[ks].v, false, bf.v, (short)0, acc, false, false);
            }
#pragma unroll
            for (int r = 0; r < 8; ++r) {
                int i = wm * 16 + r + msel;
                int c = h2 * 256 + cl + lr;
                size_t pix = ((size_t)(b * 64) + (mode ? f : i)) * 64 + (mode ? i : f);
                size_t o = pix * 512 + c;
                if (!mode) tmpH[o] = acc[r];
                else {
                    float val = g0 * (tmpH[o] + acc[r]) + bf2f(xin[o]);
                    oout[o] = f2bf(val);
                }
            }
        }
        __syncthreads();
    }
}

// ---------------------------------------------------------------------------
extern "C" void kernel_launch(void* const* d_in, const int* in_sizes, int n_in,
                              void* d_out, int out_size, void* d_ws, size_t ws_size,
                              hipStream_t stream) {
    const float* x       = (const float*)d_in[0];
    const float* conva_w = (const float*)d_in[1];
    const float* conva_g = (const float*)d_in[2];
    const float* conva_b = (const float*)d_in[3];
    const float* conva_m = (const float*)d_in[4];
    const float* conva_v = (const float*)d_in[5];
    const float* q_w     = (const float*)d_in[6];
    const float* q_b     = (const float*)d_in[7];
    const float* k_w     = (const float*)d_in[8];
    const float* k_b     = (const float*)d_in[9];
    const float* v_w     = (const float*)d_in[10];
    const float* v_b     = (const float*)d_in[11];
    const float* cc_gamma= (const float*)d_in[12];
    const float* convb_w = (const float*)d_in[13];
    const float* convb_g = (const float*)d_in[14];
    const float* convb_b = (const float*)d_in[15];
    const float* convb_m = (const float*)d_in[16];
    const float* convb_v = (const float*)d_in[17];
    const float* bott_w  = (const float*)d_in[18];
    const float* bott_g  = (const float*)d_in[19];
    const float* bott_b  = (const float*)d_in[20];
    const float* bott_m  = (const float*)d_in[21];
    const float* bott_v  = (const float*)d_in[22];
    const float* cls_w   = (const float*)d_in[23];
    const float* cls_b   = (const float*)d_in[24];

    char* p = (char*)d_ws;
    auto alloc = [&](size_t bytes) -> void* {
        void* r = (void*)p;
        p += (bytes + 255) & ~(size_t)255;
        return r;
    };
    typedef unsigned short bfu;
    bfu* xb  = (bfu*)alloc((size_t)NPIX * 2048 * 2);
    bfu* wA  = (bfu*)alloc((size_t)512 * 9 * 2048 * 2);
    bfu* wB  = (bfu*)alloc((size_t)512 * 9 * 512 * 2);
    bfu* wT  = (bfu*)alloc((size_t)512 * 9 * 2560 * 2);
    bfu* wQ  = (bfu*)alloc((size_t)64 * 512 * 2);
    bfu* wK  = (bfu*)alloc((size_t)64 * 512 * 2);
    bfu* wV  = (bfu*)alloc((size_t)512 * 512 * 2);
    bfu* wC  = (bfu*)alloc((size_t)104 * 512 * 2);
    float* prm = (float*)alloc(32768);
    float *sA = prm,        *bA = prm + 512;
    float *sB = prm + 1024, *bB = prm + 1536;
    float *sT = prm + 2048, *bT = prm + 2560;
    float *sQ = prm + 3072, *bQ = prm + 3136;
    float *sK = prm + 3200, *bK = prm + 3264;
    float *sV = prm + 3328, *bV = prm + 3840;
    float *sC = prm + 4352, *bC = prm + 4480;
    bfu* o1   = (bfu*)alloc((size_t)NPIX * 512 * 2);
    bfu* o2   = (bfu*)alloc((size_t)NPIX * 512 * 2);
    bfu* qb   = (bfu*)alloc((size_t)NPIX * 64 * 2);
    bfu* kb   = (bfu*)alloc((size_t)NPIX * 64 * 2);
    bfu* vb   = (bfu*)alloc((size_t)NPIX * 512 * 2);
    float* sc = (float*)alloc((size_t)NPIX * 128 * 4);
    bfu* aH   = (bfu*)alloc((size_t)NPIX * 64 * 2);
    bfu* aW   = (bfu*)alloc((size_t)NPIX * 64 * 2);
    float* tH = (float*)alloc((size_t)NPIX * 512 * 4);
    bfu* cb   = (bfu*)alloc((size_t)NPIX * 512 * 2);
    bfu* bo   = (bfu*)alloc((size_t)NPIX * 512 * 2);

    // ---- input & weight transforms ----
    cvt_nhwc<<<dim3(2048 / 32, 128, 8), 256, 0, stream>>>(x, xb, 2048);
    auto wlaunch = [&](const float* s, bfu* d, int Cout, int Ct, int KHW) {
        int total = Cout * KHW * Ct;
        wtrans<<<(total + 255) / 256, 256, 0, stream>>>(s, d, Cout, Ct, KHW);
    };
    wlaunch(conva_w, wA, 512, 2048, 9);
    wlaunch(convb_w, wB, 512, 512, 9);
    wlaunch(bott_w,  wT, 512, 2560, 9);
    wlaunch(q_w, wQ, 64, 512, 1);
    wlaunch(k_w, wK, 64, 512, 1);
    wlaunch(v_w, wV, 512, 512, 1);
    wlaunch(cls_w, wC, 104, 512, 1);
    bn_prep<<<2, 256, 0, stream>>>(conva_g, conva_b, conva_m, conva_v, sA, bA, 512);
    bn_prep<<<2, 256, 0, stream>>>(convb_g, convb_b, convb_m, convb_v, sB, bB, 512);
    bn_prep<<<2, 256, 0, stream>>>(bott_g, bott_b, bott_m, bott_v, sT, bT, 512);
    bias_prep<<<1, 64,  0, stream>>>(q_b, sQ, bQ, 64);
    bias_prep<<<1, 64,  0, stream>>>(k_b, sK, bK, 64);
    bias_prep<<<2, 256, 0, stream>>>(v_b, sV, bV, 512);
    bias_prep<<<1, 128, 0, stream>>>(cls_b, sC, bC, 104);

    // ---- conva (3x3, 2048->512, BN+ReLU) ----
    conv_gemm<<<dim3(4, 256), 256, 0, stream>>>(xb, 2048, nullptr, 0, wA, 512, 3, 3,
                                                sA, bA, 1, o1, nullptr);
    // ---- 2x criss-cross attention ----
    bfu* cin = o1;
    bfu* cout = o2;
    for (int it = 0; it < 2; ++it) {
        conv_gemm<<<dim3(1, 256), 256, 0, stream>>>(cin, 512, nullptr, 0, wQ, 64, 1, 1,
                                                    sQ, bQ, 0, qb, nullptr);
        conv_gemm<<<dim3(1, 256), 256, 0, stream>>>(cin, 512, nullptr, 0, wK, 64, 1, 1,
                                                    sK, bK, 0, kb, nullptr);
        conv_gemm<<<dim3(4, 256), 256, 0, stream>>>(cin, 512, nullptr, 0, wV, 512, 1, 1,
                                                    sV, bV, 0, vb, nullptr);
        cca_scores<<<512, 128, 0, stream>>>(qb, kb, sc, 0);
        cca_scores<<<512, 128, 0, stream>>>(qb, kb, sc, 1);
        cca_softmax<<<NPIX / 8, 256, 0, stream>>>(sc, aH, aW);
        cca_out<<<512, 256, 0, stream>>>(aH, vb, tH, nullptr, nullptr, nullptr, 0);
        cca_out<<<512, 256, 0, stream>>>(aW, vb, tH, cin, cc_gamma, cout, 1);
        bfu* t = cin; cin = cout; cout = t;     // cin now holds this iteration's output
    }
    // ---- convb (3x3, 512->512, BN+ReLU) ----
    conv_gemm<<<dim3(4, 256), 256, 0, stream>>>(cin, 512, nullptr, 0, wB, 512, 3, 3,
                                                sB, bB, 1, cb, nullptr);
    // ---- bott (3x3 on concat[x(2048), convb(512)] -> 512, BN+ReLU) ----
    conv_gemm<<<dim3(4, 256), 256, 0, stream>>>(xb, 2048, cb, 512, wT, 512, 3, 3,
                                                sT, bT, 1, bo, nullptr);
    // ---- cls (1x1, 512->104, +bias, fp32 NCHW out) ----
    conv_gemm<<<dim3(1, 256), 256, 0, stream>>>(bo, 512, nullptr, 0, wC, 104, 1, 1,
                                                sC, bC, 0, nullptr, (float*)d_out);
}
